// multi_attention_12601434047064
// MI455X (gfx1250) — matrix operand order repaired and checked by reading, not compiled, not run on hardware
//
#include <hip/hip_runtime.h>
#include <math.h>

#define BATCH  16
#define NHEAD  8
#define NGENE  2048
#define BLOCK  128               // 4 waves of 32 (wave32 on gfx1250)
#define WAVES  (BLOCK / 32)
#define GTILE  (WAVES * 16)      // 64 genes per workgroup
#define KCHUNK 32                // K-dim of v_wmma_f32_16x16x32_bf16
#define XREG   (NGENE / BLOCK)   // 16 x-values cached per thread

typedef __attribute__((ext_vector_type(16))) __bf16 v16bf;
typedef __attribute__((ext_vector_type(8)))  float  v8f;
typedef __attribute__((ext_vector_type(8)))  int    v8i;
typedef __attribute__((ext_vector_type(4)))  float  f4;
typedef __attribute__((ext_vector_type(4)))  int    i4;

#define LOG2E 1.4426950408889634f

__global__ void __launch_bounds__(BLOCK)
attn_rank1_softmax_kernel(const float* __restrict__ x,
                          const float* __restrict__ WQ,
                          const float* __restrict__ WK,
                          const float* __restrict__ WV,
                          const float* __restrict__ W0,
                          float* __restrict__ out)
{
    __shared__ alignas(16) float  sK[NGENE];     // K[b,h,:] f32 (exp operand)
    __shared__ alignas(16) __bf16 sVbf[NGENE];   // V[b,h,:] pre-converted bf16 (A operand)
    __shared__ float sred[2 * WAVES];            // per-wave max/min partials

    const int tid  = threadIdx.x;
    const int wave = tid >> 5;
    const int lane = tid & 31;
    const int n    = lane & 15;          // gene column in tile / A-matrix row id
    const int hi   = lane >> 4;          // lane half (K-group selector)

    const int b      = blockIdx.x / (NGENE / GTILE);
    const int gchunk = blockIdx.x % (NGENE / GTILE);
    const int g      = gchunk * GTILE + wave * 16 + n;   // this lane's gene

    const float xg = x[b * NGENE + g];

    // Branchless A-operand constants: row0 = 1.0(bf16 pair), row1 = V, rows 2..15 = 0
    const int onesdw = (n == 0) ? 0x3f803f80 : 0;

    const f4* sK4  = (const f4*)sK;
    const i4* sVb4 = (const i4*)sVbf;

    // cache x[b, tid + j*BLOCK] in registers across all 8 heads
    float xr[XREG];
    #pragma unroll
    for (int j = 0; j < XREG; ++j)
        xr[j] = x[b * NGENE + tid + j * BLOCK];

    float zacc = 0.0f;                                   // Σ_h W0[h] * z[b,h,g]

    for (int h = 0; h < NHEAD; ++h) {
        // issue per-head scalars/loads early so latency hides under the LDS fill
        const float w0h = W0[h];
        const float qg  = xg * WQ[h * NGENE + g];
        const float kg  = xg * WK[h * NGENE + g];
        const float vg  = xg * WV[h * NGENE + g];

        // ---- build K (f32) and V (bf16) for (b,h) in LDS; track K min/max ----
        float kmx = -INFINITY, kmn = INFINITY;
        #pragma unroll
        for (int j = 0; j < XREG; ++j) {
            const int i = tid + j * BLOCK;
            const float kv = xr[j] * WK[h * NGENE + i];
            const float vv = xr[j] * WV[h * NGENE + i];
            sK[i]   = kv;
            sVbf[i] = (__bf16)vv;
            kmx = fmaxf(kmx, kv);
            kmn = fminf(kmn, kv);
        }
        // wave32 butterfly reduce; each thread folds the 4 wave partials itself
        #pragma unroll
        for (int off = 16; off > 0; off >>= 1) {
            kmx = fmaxf(kmx, __shfl_xor(kmx, off));
            kmn = fminf(kmn, __shfl_xor(kmn, off));
        }
        if (lane == 0) { sred[wave] = kmx; sred[WAVES + wave] = kmn; }
        __syncthreads();   // also publishes sK / sVbf
        float kmax = sred[0], kmin = sred[WAVES];
        #pragma unroll
        for (int w = 1; w < WAVES; ++w) {
            kmax = fmaxf(kmax, sred[w]);
            kmin = fminf(kmin, sred[WAVES + w]);
        }

        // Fold exp->exp2: exp(qg*K - mg) = exp2(qg2*K - mg2)
        const float qg2 = qg * LOG2E;
        const float mg2 = (qg2 >= 0.0f) ? qg2 * kmax : qg2 * kmin;
        // masked-diagonal term, computed BEFORE the i-loop (off the critical tail)
        const float diag = __builtin_amdgcn_exp2f(fmaf(qg2, kg, -mg2)) * vg;

        v8f acc = {};   // D: row0 = S1 per gene-column, row1 = SV per gene-column

        #pragma unroll 2
        for (int i0 = 0; i0 < NGENE; i0 += KCHUNK) {
            // ---- B = E^T chunk (32x16 bf16). Lane owns gene column N=n.
            //      Bm[t] = E[K = i0 + 16*hi + t], 16 contiguous K's.
            const int rB = (i0 >> 2) + hi * 4;
            const f4 kq0 = sK4[rB + 0];
            const f4 kq1 = sK4[rB + 1];
            const f4 kq2 = sK4[rB + 2];
            const f4 kq3 = sK4[rB + 3];
            const float kv[16] = { kq0[0], kq0[1], kq0[2], kq0[3],
                                   kq1[0], kq1[1], kq1[2], kq1[3],
                                   kq2[0], kq2[1], kq2[2], kq2[3],
                                   kq3[0], kq3[1], kq3[2], kq3[3] };
            v16bf Bm;
            #pragma unroll
            for (int t = 0; t < 16; ++t)
                Bm[t] = (__bf16)__builtin_amdgcn_exp2f(fmaf(qg2, kv[t], -mg2));

            // ---- A (16x32 bf16): row0 = ones, row1 = V, rows 2..15 = 0.
            //      Am dwords 0..3 = bf16 pairs K = i0+16*hi .. +8,
            //      Am dwords 4..7 = bf16 pairs K = i0+16*hi+8 .. +8.
            //      One v_cndmask_b32 per dword against hoisted constants.
            const int rA = ((i0 + 16 * hi) >> 3);    // i4 index (8 bf16 per i4)
            const i4 va = sVb4[rA];
            const i4 vb = sVb4[rA + 1];
            v8i ai;
            ai[0] = (n == 1) ? va[0] : onesdw;
            ai[1] = (n == 1) ? va[1] : onesdw;
            ai[2] = (n == 1) ? va[2] : onesdw;
            ai[3] = (n == 1) ? va[3] : onesdw;
            ai[4] = (n == 1) ? vb[0] : onesdw;
            ai[5] = (n == 1) ? vb[1] : onesdw;
            ai[6] = (n == 1) ? vb[2] : onesdw;
            ai[7] = (n == 1) ? vb[3] : onesdw;
            const v16bf Am = __builtin_bit_cast(v16bf, ai);

            // D = A x B + C  (v_wmma_f32_16x16x32_bf16)
            acc = __builtin_amdgcn_wmma_f32_16x16x32_bf16(
                false, Am, false, Bm, (short)0, acc, false, false);
        }

        // ---- epilogue: lanes 0-15 hold (M=0,N=n)=S1 in acc[0], (M=1,N=n)=SV in acc[1]
        // fast reciprocal (v_rcp_f32) instead of IEEE division sequence
        const float z = (acc[1] - diag) * __builtin_amdgcn_rcpf(acc[0]);
        zacc = fmaf(w0h, z, zacc);

        __syncthreads();   // protect sK/sVbf/sred before next head overwrites them
    }

    if (hi == 0)
        out[b * NGENE + g] = zacc;   // each (b,g) written exactly once
}

extern "C" void kernel_launch(void* const* d_in, const int* in_sizes, int n_in,
                              void* d_out, int out_size, void* d_ws, size_t ws_size,
                              hipStream_t stream) {
    (void)in_sizes; (void)n_in; (void)out_size; (void)d_ws; (void)ws_size;
    const float* x  = (const float*)d_in[0];
    const float* WQ = (const float*)d_in[1];
    const float* WK = (const float*)d_in[2];
    const float* WV = (const float*)d_in[3];
    const float* W0 = (const float*)d_in[4];
    float* out = (float*)d_out;

    dim3 grid(BATCH * (NGENE / GTILE));   // 16 * 32 = 512 workgroups
    dim3 block(BLOCK);                    // 4 waves
    attn_rank1_softmax_kernel<<<grid, block, 0, stream>>>(x, WQ, WK, WV, W0, out);
}
